// FTLSTM_30623116820666
// MI455X (gfx1250) — matrix operand order repaired and checked by reading, not compiled, not run on hardware
//
#include <hip/hip_runtime.h>

// ---------------------------------------------------------------------------
// FT-LSTM (bidirectional time-aware LSTM) for MI455X / gfx1250.
//   B=64, S=1024, D=256, H=256.
// Strategy:
//   K1 pack  : weights -> f16, WMMA B-operand lane-major layout; bias concat.
//   K2 gx    : GX[s][b][4H] = x[b,s,:] @ [Wi|Wf|Wog|Wc]  (f16, in d_ws, 128MB)
//              shared by fwd+bwd directions. WMMA f32_16x16x32_f16.
//   K3 scan  : persistent kernel, grid=<2 blocks (fwd/bwd)>, 1024 thr (32 waves)
//              per block = 1 WGP per direction. h/c state in LDS, s_barrier
//              sync (2/step). 5 GEMM chains per tile (i,f,o,Cg gates + decomp),
//              GX folded in as WMMA C-init. Elementwise cell update in regs.
// Workspace layout (bytes):
//   [0,        512K)  Wxpk   256x1024 f16 (packed)
//   [512K,     1M  )  Upk    256x1024 f16 (packed)
//   [1M,    1.125M )  Wdpk   256x256  f16 (packed)
//   [1179648, +5K  )  bias   1280 f32  (bi|bf|bog|bc|b_decomp)
//   [1184768, +128M)  GX     S*B*4H f16
// ---------------------------------------------------------------------------

typedef __attribute__((ext_vector_type(16))) _Float16 v16h;
typedef __attribute__((ext_vector_type(8)))  _Float16 v8h;
typedef __attribute__((ext_vector_type(8)))  float    v8f;

#define B_   64
#define S_   1024
#define D_   256
#define H_   256
#define G4   1024      // 4*H
#define HSTR 264       // padded LDS row stride (f16) -> bank-conflict free b128

struct Scalars {
    const float *wd1, *wd2, *wd3, *a, *b, *m, *k, *d, *n;
};

// ---------------------------------------------------------------------------
// K1: pack weights into WMMA B-operand layout.
// Packed element index e = ((nt*8 + ks)*32 + lane)*16 + v maps to
//   n = nt*16 + (lane&15),  K = ks*32 + (lane>>4)*8 + (v&7) + (v>>3)*16,
// so each lane's 16 halves for one (nt,ks) block are contiguous (32B load).
// ---------------------------------------------------------------------------
__global__ __launch_bounds__(256) void pack_kernel(
    const float* Wi, const float* Ui, const float* Wf, const float* Uf,
    const float* Wog, const float* Uog, const float* Wc, const float* Uc,
    const float* Wd,
    const float* bi, const float* bf, const float* bog, const float* bc,
    const float* bd,
    _Float16* Wxpk, _Float16* Upk, _Float16* Wdpk, float* biasws)
{
    const int idx = blockIdx.x * blockDim.x + threadIdx.x;   // 0 .. 262143
    const float* Wg[4] = {Wi, Wf, Wog, Wc};
    const float* Ug[4] = {Ui, Uf, Uog, Uc};

    const int v    = idx & 15;
    const int lane = (idx >> 4) & 31;
    const int ks   = (idx >> 9) & 7;
    const int nt   = idx >> 12;                 // 0..63
    const int n    = nt * 16 + (lane & 15);     // 0..1023
    const int K    = ks * 32 + (lane >> 4) * 8 + (v & 7) + (v >> 3) * 16;  // 0..255
    const int g    = n >> 8;
    const int col  = n & 255;

    Wxpk[idx] = (_Float16)Wg[g][K * H_ + col];
    Upk[idx]  = (_Float16)Ug[g][K * H_ + col];
    if (nt < 16) Wdpk[idx] = (_Float16)Wd[K * H_ + n];

    if (idx < 1280) {
        const float* bg[4] = {bi, bf, bog, bc};
        biasws[idx] = (idx < 1024) ? bg[idx >> 8][idx & 255] : bd[idx - 1024];
    }
}

// ---------------------------------------------------------------------------
// K2: GX[s*64+b][4H] = x[b,s,:] @ Wcat.   65536x1024x256 GEMM, f16 out.
// 8 waves/block, 1 (16x16) tile per wave, 8 K-step WMMA chain.
// ---------------------------------------------------------------------------
__global__ __launch_bounds__(256) void gx_kernel(
    const float* __restrict__ x, const _Float16* __restrict__ Wxpk,
    _Float16* __restrict__ GX)
{
    const int wave = threadIdx.x >> 5;
    const int lane = threadIdx.x & 31;
    const int ln   = lane & 15;
    const int kh   = lane >> 4;

    const int tile = blockIdx.x * 8 + wave;   // 0 .. 262143
    const int mt   = tile >> 6;               // 0 .. 4095 (row tile over B*S)
    const int nt   = tile & 63;               // 0 .. 63   (col tile over 4H)
    const int s    = mt >> 2;
    const int b0   = (mt & 3) * 16;

    // A row this lane supplies: batch b0+ln at sequence position s.
    const float* xrow = x + ((size_t)(b0 + ln) * S_ + s) * D_;

    v8f acc = {};
#pragma unroll
    for (int ks = 0; ks < 8; ++ks) {
        // A operand: f32 -> f16 on the fly, matching the 16-bit A layout
        // (lanes 0-15: K = ks*32 + {0..7, 16..23}; lanes 16-31: +8).
        const float* p0 = xrow + ks * 32 + kh * 8;
        const float* p1 = p0 + 16;
        v16h A;
#pragma unroll
        for (int i = 0; i < 8; ++i) {
            A[i]     = (_Float16)p0[i];
            A[8 + i] = (_Float16)p1[i];
        }
        const v16h Bv = *(const v16h*)(Wxpk + ((size_t)(nt * 8 + ks) * 32 + lane) * 16);
        acc = __builtin_amdgcn_wmma_f32_16x16x32_f16(false, A, false, Bv,
                                                     (short)0, acc, false, false);
    }
#pragma unroll
    for (int vr = 0; vr < 8; ++vr) {
        const int r = mt * 16 + vr + 8 * kh;            // = s*64 + b
        GX[(size_t)r * G4 + nt * 16 + ln] = (_Float16)acc[vr];
    }
}

// ---------------------------------------------------------------------------
// A-operand fetch from padded LDS: two 16B ds loads merged to v16h.
// ---------------------------------------------------------------------------
__device__ __forceinline__ v16h lds_a(const _Float16* base, int off) {
    const v8h a0 = *(const v8h*)(base + off);
    const v8h a1 = *(const v8h*)(base + off + 16);
    return __builtin_shufflevector(a0, a1, 0, 1, 2, 3, 4, 5, 6, 7,
                                   8, 9, 10, 11, 12, 13, 14, 15);
}

// ---------------------------------------------------------------------------
// K3: the sequential scan. blockIdx.x = direction (0 fwd, 1 bwd).
// 32 waves; wave w owns output tiles q=w and q=w+32, q -> (mtile=q>>4, jt=q&15).
// Per step: 5 WMMA chains/tile (4 gates from h@U, decomp from c@Wd), GX as
// C-init, then in-register cell update; h/c written back to LDS; 2 barriers.
// ---------------------------------------------------------------------------
__global__ __launch_bounds__(1024) void scan_kernel(
    const float* __restrict__ times, const _Float16* __restrict__ GX,
    const _Float16* __restrict__ Upk, const _Float16* __restrict__ Wdpk,
    const float* __restrict__ biasg, float* __restrict__ out, Scalars sp)
{
    extern __shared__ char smem[];
    _Float16* h16 = (_Float16*)smem;              // 64*HSTR halves
    _Float16* c16 = h16 + 64 * HSTR;              // 64*HSTR halves
    float*    c32 = (float*)(c16 + 64 * HSTR);    // 64*256
    float*    Tl  = c32 + 64 * 256;               // 64
    float*    bl  = Tl + 64;                      // 1280

    const int tid  = threadIdx.x;
    const int dir  = blockIdx.x;
    const int wave = tid >> 5;
    const int lane = tid & 31;
    const int ln   = lane & 15;
    const int kh   = lane >> 4;

    for (int i = tid; i < 64 * HSTR; i += 1024) {
        h16[i] = (_Float16)0.f;
        c16[i] = (_Float16)0.f;
    }
    for (int i = tid; i < 64 * 256; i += 1024) c32[i] = 0.f;
    for (int i = tid; i < 1280;     i += 1024) bl[i]  = biasg[i];
    __syncthreads();

    const float Pw1 = *sp.wd1, Pw2 = *sp.wd2, Pw3 = *sp.wd3;
    const float Pa = *sp.a, Pb = *sp.b, Pm = *sp.m;
    const float Pk = *sp.k, Pd = *sp.d, Pn = *sp.n;
    const size_t HL = (size_t)B_ * S_ * (2 * H_);

    for (int s = 0; s < S_; ++s) {
        const int sx = dir ? (S_ - 1 - s) : s;                 // x / GX index
        const int st = dir ? (s == 0 ? 0 : S_ - s) : s;        // time index

        if (tid < 64) {   // waves 0-1: elapsed-time decay T[b]
            const float t  = times[(size_t)tid * S_ + st];
            const float T1 = 1.f / (Pa * powf(t, Pb));
            const float T2 = Pk - Pm * t;
            const float T3 = 1.f / (1.f + powf(t / Pd, Pn));
            const float T  = Pw1 * T1 + Pw2 * T2 + Pw3 * T3;
            Tl[tid] = fminf(fmaxf(T, 0.f), 1.f);
        }

        const _Float16* gxs = GX + (size_t)sx * B_ * G4;
        v8f acc[2][5];

#pragma unroll
        for (int jj = 0; jj < 2; ++jj) {
            const int q   = wave + jj * 32;
            const int mtl = q >> 4;
            const int jt  = q & 15;
            const int j   = jt * 16 + ln;

            // C-init: 4 gate tiles from precomputed x@W; decomp starts at 0.
#pragma unroll
            for (int vr = 0; vr < 8; ++vr) {
                const size_t ro = (size_t)(mtl * 16 + vr + 8 * kh) * G4 + j;
                acc[jj][0][vr] = (float)gxs[ro];
                acc[jj][1][vr] = (float)gxs[ro + H_];
                acc[jj][2][vr] = (float)gxs[ro + 2 * H_];
                acc[jj][3][vr] = (float)gxs[ro + 3 * H_];
                acc[jj][4][vr] = 0.f;
            }

#pragma unroll
            for (int ks = 0; ks < 8; ++ks) {
                const int rb = (mtl * 16 + ln) * HSTR + ks * 32 + kh * 8;
                const v16h Ah = lds_a(h16, rb);
                const v16h Ac = lds_a(c16, rb);
#pragma unroll
                for (int g = 0; g < 4; ++g) {
                    const v16h Bv = *(const v16h*)(
                        Upk + ((size_t)((g * 16 + jt) * 8 + ks) * 32 + lane) * 16);
                    acc[jj][g] = __builtin_amdgcn_wmma_f32_16x16x32_f16(
                        false, Ah, false, Bv, (short)0, acc[jj][g], false, false);
                }
                const v16h Bd = *(const v16h*)(
                    Wdpk + ((size_t)(jt * 8 + ks) * 32 + lane) * 16);
                acc[jj][4] = __builtin_amdgcn_wmma_f32_16x16x32_f16(
                    false, Ac, false, Bd, (short)0, acc[jj][4], false, false);
            }
        }
        __syncthreads();   // GEMM LDS reads + T done before state rewrite

#pragma unroll
        for (int jj = 0; jj < 2; ++jj) {
            const int q   = wave + jj * 32;
            const int mtl = q >> 4;
            const int jt  = q & 15;
            const int j   = jt * 16 + ln;
#pragma unroll
            for (int vr = 0; vr < 8; ++vr) {
                const int   b  = mtl * 16 + vr + 8 * kh;
                const float T  = Tl[b];
                const float cp = c32[b * 256 + j];
                const float cs = tanhf(acc[jj][4][vr] + bl[1024 + j]);
                const float cb = cp - cs + T * cs;
                const float iv = 1.f / (1.f + expf(-(acc[jj][0][vr] + bl[j])));
                const float fv = 1.f / (1.f + expf(-(acc[jj][1][vr] + bl[256 + j])));
                const float ov = 1.f / (1.f + expf(-(acc[jj][2][vr] + bl[512 + j])));
                const float gv = 1.f / (1.f + expf(-(acc[jj][3][vr] + bl[768 + j])));
                const float ct = fv * cb + iv * gv;
                const float hv = ov * tanhf(ct);

                c32[b * 256 + j]  = ct;
                c16[b * HSTR + j] = (_Float16)ct;
                h16[b * HSTR + j] = (_Float16)hv;
                out[((size_t)b * S_ + s) * (2 * H_) + dir * H_ + j] = hv;
                if (s == S_ - 1) {   // h_last / c_last
                    out[HL + (size_t)b * (2 * H_) + dir * H_ + j] = hv;
                    out[HL + (size_t)B_ * (2 * H_) + (size_t)b * (2 * H_) + dir * H_ + j] = ct;
                }
            }
        }
        __syncthreads();   // state visible before next step's GEMM
    }
}

// ---------------------------------------------------------------------------
extern "C" void kernel_launch(void* const* d_in, const int* in_sizes, int n_in,
                              void* d_out, int out_size, void* d_ws, size_t ws_size,
                              hipStream_t stream)
{
    const float* inputs = (const float*)d_in[0];
    const float* times  = (const float*)d_in[1];
    const float* Wi  = (const float*)d_in[2];
    const float* Ui  = (const float*)d_in[3];
    const float* bi  = (const float*)d_in[4];
    const float* Wf  = (const float*)d_in[5];
    const float* Uf  = (const float*)d_in[6];
    const float* bf  = (const float*)d_in[7];
    const float* Wog = (const float*)d_in[8];
    const float* Uog = (const float*)d_in[9];
    const float* bog = (const float*)d_in[10];
    const float* Wc  = (const float*)d_in[11];
    const float* Uc  = (const float*)d_in[12];
    const float* bc  = (const float*)d_in[13];
    const float* Wd  = (const float*)d_in[14];
    const float* bd  = (const float*)d_in[15];
    Scalars sp{(const float*)d_in[16], (const float*)d_in[17],
               (const float*)d_in[18], (const float*)d_in[19],
               (const float*)d_in[20], (const float*)d_in[21],
               (const float*)d_in[22], (const float*)d_in[23],
               (const float*)d_in[24]};

    char* ws = (char*)d_ws;
    _Float16* Wxpk   = (_Float16*)(ws + 0);
    _Float16* Upk    = (_Float16*)(ws + 524288);
    _Float16* Wdpk   = (_Float16*)(ws + 1048576);
    float*    biasws = (float*)   (ws + 1179648);
    _Float16* GX     = (_Float16*)(ws + 1184768);   // 128 MB

    pack_kernel<<<1024, 256, 0, stream>>>(Wi, Ui, Wf, Uf, Wog, Uog, Wc, Uc, Wd,
                                          bi, bf, bog, bc, bd,
                                          Wxpk, Upk, Wdpk, biasws);
    gx_kernel<<<32768, 256, 0, stream>>>(inputs, Wxpk, GX);

    const size_t smem = (size_t)(64 * HSTR * 2 * sizeof(_Float16)  // h16+c16
                                 + 64 * 256 * sizeof(float)        // c32
                                 + 64 * sizeof(float)              // T
                                 + 1280 * sizeof(float));          // bias
    scan_kernel<<<2, 1024, smem, stream>>>(times, GX, Upk, Wdpk, biasws,
                                           (float*)d_out, sp);
}
